// genformer_72945724555586
// MI455X (gfx1250) — matrix operand order repaired and checked by reading, not compiled, not run on hardware
//
#include <hip/hip_runtime.h>
#include <hip/hip_bf16.h>
#include <math.h>

// ---------------------------------------------------------------------------
// MI455X (gfx1250) implementation: all large contractions go through
// v_wmma_f32_16x16x32_bf16. Activations are kept in bf16 so the whole
// inter-kernel working set (< ~172MB) lives in the 192MB L2.
// A-tiles are staged global->LDS with the CDNA5 async-DMA path (ASYNCcnt +
// s_wait_asynccnt) and double-buffered in LDS so the copy of chunk k+1
// overlaps the WMMAs of chunk k.
// ---------------------------------------------------------------------------

typedef __bf16 bf16;
typedef __attribute__((ext_vector_type(16))) __bf16 v16bf;
typedef __attribute__((ext_vector_type(8)))  float  v8f;
typedef int gf_v4i __attribute__((vector_size(16)));

union AF {
    v16bf v;
    __bf16 h[16];
    uint4  q[2];
};

union BV {
    uint4  q;
    __bf16 h[8];
};

#if defined(__has_builtin)
# if __has_builtin(__builtin_amdgcn_global_load_async_to_lds_b128) && \
     __has_builtin(__builtin_amdgcn_s_wait_asynccnt)
#  define GF_ASYNC 1
# endif
#endif

__device__ __forceinline__ void gf_stage_b128(const bf16* g, bf16* l) {
#ifdef GF_ASYNC
    __builtin_amdgcn_global_load_async_to_lds_b128(
        (__attribute__((address_space(1))) gf_v4i*)(void*)g,
        (__attribute__((address_space(3))) gf_v4i*)(void*)l,
        0, 0);
#else
    *reinterpret_cast<uint4*>(l) = *reinterpret_cast<const uint4*>(g);
#endif
}

__device__ __forceinline__ void gf_wait_stage() {
#ifdef GF_ASYNC
    __builtin_amdgcn_s_wait_asynccnt(0);
#endif
}

__device__ __forceinline__ float gf_gelu(float x) {
    return 0.5f * x * (1.0f + erff(x * 0.70710678118f));
}

// ---------------------------------------------------------------------------
// Generic bf16 WMMA GEMM / implicit-GEMM conv1d.
//   C[M,N] = act(alpha * (A ∘ Bw) + bias) + resid
//   convW == 1 : A is [M, lda], plain GEMM, K arbitrary (tail predicated)
//   convW  > 1 : A is [M(rows=L), cin] activations; logical K = convW*cin,
//                gather row r = m + w - (convW-1)/2  (requires cin%32==0)
//   Bw is [K, N] row-major bf16 (matches the reference (W,Cin,Cout) layout).
//   act: 0 none, 1 exact GELU, 2 relu+STAB (Performer feature map)
//   storeT: write C[n*ldc + m] instead (produces k^T for the kv einsum)
// Block: 256 threads = 8 waves; tile 128(M) x 64(N); wave tile 32x32
//        = 2 A-frags x 2 B-frags x 4 WMMA per 32-wide K chunk.
// Both A and B(transposed) fragments are two 16B ds_load_b128 per the
// ISA 7.12.2 bf16 register layouts. LDS is double-buffered: chunk k+1's
// async A-copy and B global load are issued before chunk k's WMMAs, and
// ASYNCcnt is drained only after compute.
// ---------------------------------------------------------------------------
#define TM 128
#define TN 64
#define KB 32
#define LDT_S 40   // LDS row stride (bf16) => 80B rows, 16B-aligned K-runs

__global__ __launch_bounds__(256) void gf_wmma_gemm(
    const bf16* __restrict__ A, int lda,
    const bf16* __restrict__ Bw, int ldb,
    const float* __restrict__ bias,
    const bf16* __restrict__ resid, int ldr,
    bf16* __restrict__ C, int ldc,
    int M, int N, int K,
    int convW, int cin,
    float alpha, int act, int storeT)
{
    __shared__ __align__(16) bf16 As [2][TM * LDT_S];
    __shared__ __align__(16) bf16 BsT[2][TN * LDT_S];

    const int tid  = threadIdx.x;
    const int lane = tid & 31;
    const int wav  = tid >> 5;
    const int wm   = (wav >> 1) << 5;   // 0,32,64,96
    const int wn   = (wav & 1) << 5;    // 0,32
    const int m0   = blockIdx.y * TM;
    const int n0   = blockIdx.x * TN;
    const int pad  = (convW - 1) >> 1;
    const int kr   = tid >> 3;          // B stage: K row within chunk (0..31)
    const int nv   = (tid & 7) << 3;    // B stage: N offset within tile (0..56)

    // ---- staging helpers --------------------------------------------------
    auto stageA = [&](int k0, int buf) {
        if (convW > 1) {
            // chunk stays inside one tap since cin%32==0 and k0%32==0
            int w  = k0 / cin;
            int c0 = k0 - w * cin;
            int ro = w - pad;
            for (int i = tid; i < TM * 4; i += 256) {
                int mr = i >> 2;
                int kv = (i & 3) << 3;
                int r  = m0 + mr + ro;
                bf16* dst = &As[buf][mr * LDT_S + kv];
                if (r >= 0 && r < M)
                    gf_stage_b128(&A[(size_t)r * lda + c0 + kv], dst);
                else
                    *reinterpret_cast<uint4*>(dst) = make_uint4(0u, 0u, 0u, 0u);
            }
        } else if (((lda & 7) == 0) && (k0 + KB) <= K) {
            for (int i = tid; i < TM * 4; i += 256) {
                int mr = i >> 2;
                int kv = (i & 3) << 3;
                int m  = m0 + mr;
                bf16* dst = &As[buf][mr * LDT_S + kv];
                if (m < M)
                    gf_stage_b128(&A[(size_t)m * lda + k0 + kv], dst);
                else
                    *reinterpret_cast<uint4*>(dst) = make_uint4(0u, 0u, 0u, 0u);
            }
        } else {
            for (int i = tid; i < TM * KB; i += 256) {
                int mr = i >> 5;
                int kk = i & 31;
                int m  = m0 + mr;
                int k  = k0 + kk;
                bf16 v = (bf16)0.0f;
                if (m < M && k < K) v = A[(size_t)m * lda + k];
                As[buf][mr * LDT_S + kk] = v;
            }
        }
    };
    auto loadB = [&](int k0, BV& bv) {
        int k = k0 + kr;
        int n = n0 + nv;
        if (k < K && ((ldb & 7) == 0) && (n + 8) <= N) {
            bv.q = *reinterpret_cast<const uint4*>(&Bw[(size_t)k * ldb + n]);
        } else {
            for (int j = 0; j < 8; ++j) {
                int nn = n + j;
                bv.h[j] = (k < K && nn < N) ? Bw[(size_t)k * ldb + nn] : (bf16)0.0f;
            }
        }
        // prefetch the chunk after this one into GL2 (global_prefetch_b8)
        if (k0 + KB < K) {
            size_t pk = (size_t)((k0 + KB + kr) < K ? (k0 + KB + kr) : (K - 1));
            __builtin_prefetch(&Bw[pk * ldb + n0], 0, 1);
        }
    };
    auto storeB = [&](int buf, const BV& bv) {
        for (int j = 0; j < 8; ++j)
            BsT[buf][(nv + j) * LDT_S + kr] = bv.h[j];
    };

    // ---- prologue: stage chunk 0 into buffer 0 ----------------------------
    {
        BV bv;
        loadB(0, bv);
        stageA(0, 0);
        storeB(0, bv);
        gf_wait_stage();
        __syncthreads();
    }

    v8f acc[2][2] = {};
    const int nk = (K + KB - 1) / KB;
    int buf = 0;

    for (int c = 0; c < nk; ++c) {
        const bool hasNext = (c + 1) < nk;
        BV bv;
        if (hasNext) {
            loadB((c + 1) * KB, bv);        // global load issued early
            stageA((c + 1) * KB, buf ^ 1);  // async DMA into other buffer
        }

        // ---------------- fragments: 2x ds_load_b128 each ------------------
        AF af[2];
        {
            int am = wm + (lane & 15);
            int kb = (lane >> 4) << 3;           // 0 or 8
            for (int i = 0; i < 2; ++i) {
                af[i].q[0] = *reinterpret_cast<const uint4*>(&As[buf][(am + 16 * i) * LDT_S + kb]);
                af[i].q[1] = *reinterpret_cast<const uint4*>(&As[buf][(am + 16 * i) * LDT_S + kb + 16]);
            }
        }
        AF bfr[2];
        {
            int bn  = wn + (lane & 15);
            int bkb = (lane >> 4) << 3;          // 0 or 8
            for (int j = 0; j < 2; ++j) {
                bfr[j].q[0] = *reinterpret_cast<const uint4*>(&BsT[buf][(bn + 16 * j) * LDT_S + bkb]);
                bfr[j].q[1] = *reinterpret_cast<const uint4*>(&BsT[buf][(bn + 16 * j) * LDT_S + bkb + 16]);
            }
        }
        for (int i = 0; i < 2; ++i)
            for (int j = 0; j < 2; ++j)
                acc[i][j] = __builtin_amdgcn_wmma_f32_16x16x32_bf16(
                    false, af[i].v, false, bfr[j].v, (short)0, acc[i][j], false, false);

        if (hasNext) storeB(buf ^ 1, bv);   // transpose stores after compute
        gf_wait_stage();                    // next chunk's DMA done (overlapped)
        __syncthreads();
        buf ^= 1;
    }

    // ---------------- epilogue ----------------
    for (int i = 0; i < 2; ++i)
        for (int j = 0; j < 2; ++j)
            for (int r = 0; r < 8; ++r) {
                int m = m0 + wm + 16 * i + r + ((lane >> 4) << 3);
                int n = n0 + wn + 16 * j + (lane & 15);
                if (m < M && n < N) {
                    float v = alpha * acc[i][j][r];
                    if (bias) v += bias[n];
                    if (act == 1)      v = gf_gelu(v);
                    else if (act == 2) v = fmaxf(v, 0.0f) + 0.001f;
                    if (resid) v += (float)resid[(size_t)m * ldr + n];
                    if (storeT) C[(size_t)n * ldc + m] = (bf16)v;
                    else        C[(size_t)m * ldc + n] = (bf16)v;
                }
            }
}

// ---------------------------------------------------------------------------
// Elementwise / small kernels
// ---------------------------------------------------------------------------
__global__ void gf_f32_to_bf16(const float* __restrict__ s, bf16* __restrict__ d, long long n) {
    long long i = (long long)blockIdx.x * blockDim.x + threadIdx.x;
    if (i < n) d[i] = (bf16)s[i];
}

__global__ void gf_transpose_f32_bf16(const float* __restrict__ s, bf16* __restrict__ d, int R, int Cc) {
    long long i = (long long)blockIdx.x * blockDim.x + threadIdx.x;
    if (i >= (long long)R * Cc) return;
    int r = (int)(i / Cc), c = (int)(i % Cc);
    d[(size_t)c * R + r] = (bf16)s[i];
}

__global__ void gf_bn_gelu(const bf16* __restrict__ x, const float* __restrict__ g,
                           const float* __restrict__ b, const float* __restrict__ mu,
                           const float* __restrict__ var, bf16* __restrict__ y,
                           long long n, int C) {
    long long i = (long long)blockIdx.x * blockDim.x + threadIdx.x;
    if (i >= n) return;
    int c = (int)(i % C);
    float v = ((float)x[i] - mu[c]) * rsqrtf(var[c] + 0.001f) * g[c] + b[c];
    y[i] = (bf16)gf_gelu(v);
}

__global__ void gf_stem_conv(const float* __restrict__ in, const float* __restrict__ w,
                             const float* __restrict__ b, bf16* __restrict__ out,
                             int L, int cin, int cout, int W) {
    long long i = (long long)blockIdx.x * blockDim.x + threadIdx.x;
    if (i >= (long long)L * cout) return;
    int co = (int)(i % cout);
    int l  = (int)(i / cout);
    int pad = (W - 1) >> 1;
    float acc = b[co];
    for (int ww = 0; ww < W; ++ww) {
        int r = l + ww - pad;
        if (r < 0 || r >= L) continue;
        for (int ci = 0; ci < cin; ++ci)
            acc += in[(size_t)r * cin + ci] * w[((size_t)ww * cin + ci) * cout + co];
    }
    out[i] = (bf16)acc;
}

__global__ void gf_maxpool(const bf16* __restrict__ in, bf16* __restrict__ out,
                           int Lout, int C, int p) {
    long long i = (long long)blockIdx.x * blockDim.x + threadIdx.x;
    if (i >= (long long)Lout * C) return;
    int c  = (int)(i % C);
    int lo = (int)(i / C);
    float m = -3.0e38f;
    for (int q = 0; q < p; ++q)
        m = fmaxf(m, (float)in[((size_t)lo * p + q) * C + c]);
    out[i] = (bf16)m;
}

__global__ void gf_ln(const bf16* __restrict__ x, const float* __restrict__ g,
                      const float* __restrict__ b, bf16* __restrict__ y, int D) {
    __shared__ float sh[256];
    int row = blockIdx.x, tid = threadIdx.x;
    const bf16* xr = x + (size_t)row * D;
    float s = 0.f;
    for (int d = tid; d < D; d += 256) s += (float)xr[d];
    sh[tid] = s; __syncthreads();
    for (int o = 128; o > 0; o >>= 1) { if (tid < o) sh[tid] += sh[tid + o]; __syncthreads(); }
    float mu = sh[0] / D; __syncthreads();
    float v = 0.f;
    for (int d = tid; d < D; d += 256) { float t = (float)xr[d] - mu; v += t * t; }
    sh[tid] = v; __syncthreads();
    for (int o = 128; o > 0; o >>= 1) { if (tid < o) sh[tid] += sh[tid + o]; __syncthreads(); }
    float rstd = rsqrtf(sh[0] / D + 0.001f);
    bf16* yr = y + (size_t)row * D;
    for (int d = tid; d < D; d += 256)
        yr[d] = (bf16)(((float)xr[d] - mu) * rstd * g[d] + b[d]);
}

__global__ void gf_motif(const float* __restrict__ mot, const float* __restrict__ w1,
                         const float* __restrict__ b1, const float* __restrict__ w2,
                         const float* __restrict__ b2, float* __restrict__ mo) {
    __shared__ float m1[32];
    int t = threadIdx.x;
    if (t < 32) {
        float a = b1[t];
        for (int i = 0; i < 693; ++i) a += mot[i] * w1[i * 32 + t];
        m1[t] = gf_gelu(a);
    }
    __syncthreads();
    if (t < 8) {
        float a = b2[t];
        for (int i = 0; i < 32; ++i) a += m1[i] * w2[i * 8 + t];
        mo[t] = a;
    }
}

__global__ void gf_concat(const bf16* __restrict__ xs, const bf16* __restrict__ aa,
                          const float* __restrict__ mo, bf16* __restrict__ t) {
    long long i = (long long)blockIdx.x * blockDim.x + threadIdx.x;
    if (i >= 512LL * 1096) return;
    int c = (int)(i % 1096), s = (int)(i / 1096);
    bf16 v;
    if (c < 1024)      v = xs[(size_t)s * 1024 + c];
    else if (c < 1088) v = aa[(size_t)s * 64 + (c - 1024)];
    else               v = (bf16)mo[c - 1088];
    t[i] = v;
}

// rotary + head split: in [512,1096] -> out [4][512][274]
__global__ void gf_rotary_split(const bf16* __restrict__ in, bf16* __restrict__ out, int S) {
    long long i = (long long)blockIdx.x * blockDim.x + threadIdx.x;
    if (i >= (long long)S * 4 * 137) return;
    int j = (int)(i % 137);
    int h = (int)((i / 137) % 4);
    int s = (int)(i / (137 * 4));
    float x1 = (float)in[(size_t)s * 1096 + h * 274 + j];
    float x2 = (float)in[(size_t)s * 1096 + h * 274 + 137 + j];
    float inv = __powf(10000.0f, -(float)j / 137.0f);
    float ang = (float)s * inv;
    float sn = __sinf(ang), cs = __cosf(ang);
    bf16* o = out + ((size_t)h * S + s) * 274;
    o[j]       = (bf16)(x1 * cs - x2 * sn);
    o[137 + j] = (bf16)(x1 * sn + x2 * cs);
}

__global__ void gf_split_heads(const bf16* __restrict__ in, bf16* __restrict__ out, int S) {
    long long i = (long long)blockIdx.x * blockDim.x + threadIdx.x;
    if (i >= (long long)S * 4 * 274) return;
    int d = (int)(i % 274);
    int h = (int)((i / 274) % 4);
    int s = (int)(i / (274 * 4));
    out[((size_t)h * S + s) * 274 + d] = in[(size_t)s * 1096 + h * 274 + d];
}

// ksum[h][m] = sum_s kpT[h][m][s]
__global__ void gf_rowsum(const bf16* __restrict__ kpT, float* __restrict__ ksum) {
    int i = blockIdx.x * blockDim.x + threadIdx.x;
    if (i >= 4 * 256) return;
    const bf16* r = kpT + (size_t)i * 512;
    float s = 0.f;
    for (int t = 0; t < 512; ++t) s += (float)r[t];
    ksum[i] = s;
}

// denom[h][s] = sum_m qp[h][s][m] * ksum[h][m]
__global__ void gf_denom(const bf16* __restrict__ qp, const float* __restrict__ ksum,
                         float* __restrict__ denom) {
    int i = blockIdx.x * blockDim.x + threadIdx.x;
    if (i >= 4 * 512) return;
    int h = i >> 9, s = i & 511;
    const bf16* r = qp + ((size_t)h * 512 + s) * 256;
    const float* ks = ksum + h * 256;
    float a = 0.f;
    for (int m = 0; m < 256; ++m) a += (float)r[m] * ks[m];
    denom[i] = a;
}

// merged[s][h*274+d] = oh[h][s][d] / denom[h][s]
__global__ void gf_merge(const bf16* __restrict__ oh, const float* __restrict__ denom,
                         bf16* __restrict__ merged) {
    long long i = (long long)blockIdx.x * blockDim.x + threadIdx.x;
    if (i >= 512LL * 1096) return;
    int c = (int)(i % 1096), s = (int)(i / 1096);
    int h = c / 274, d = c % 274;
    float v = (float)oh[((size_t)h * 512 + s) * 274 + d] / denom[h * 512 + s];
    merged[i] = (bf16)v;
}

// per output position: softplus(fb[s,:] . pw + pb), trimmed to s in [2,510)
__global__ void gf_final(const bf16* __restrict__ fb, const float* __restrict__ pw,
                         const float* __restrict__ pb, float* __restrict__ out) {
    __shared__ float sh[256];
    int sp = blockIdx.x, s = sp + 2, t = threadIdx.x;
    float a = 0.f;
    for (int d = t; d < 2048; d += 256) a += (float)fb[(size_t)s * 2048 + d] * pw[d];
    sh[t] = a; __syncthreads();
    for (int o = 128; o > 0; o >>= 1) { if (t < o) sh[t] += sh[t + o]; __syncthreads(); }
    if (t == 0) {
        float v = sh[0] + pb[0];
        out[sp] = (v > 20.0f) ? v : log1pf(expf(v));
    }
}

// ---------------------------------------------------------------------------
// Orchestration
// ---------------------------------------------------------------------------
extern "C" void kernel_launch(void* const* d_in, const int* in_sizes, int n_in,
                              void* d_out, int out_size, void* d_ws, size_t ws_size,
                              hipStream_t stream) {
    (void)in_sizes; (void)n_in; (void)out_size;

    const float* seq  = (const float*)d_in[0];   // (65536,4)
    const float* atac = (const float*)d_in[1];   // (16384,1)
    const float* mot  = (const float*)d_in[2];   // (693,)
    int pi = 3;
    auto nx = [&]() -> const float* { return (const float*)d_in[pi++]; };

    // ---- workspace bump allocator (256B aligned) ----
    char* base = (char*)d_ws;
    size_t off = 0;
    auto alloc = [&](size_t bytes) -> void* {
        size_t o = (off + 255) & ~(size_t)255;
        off = o + bytes;
        return (void*)(base + o);
    };
    bf16* bA   = (bf16*)alloc(65536ull * 512 * 2);
    bf16* bB   = (bf16*)alloc(65536ull * 512 * 2);
    bf16* wscr = (bf16*)alloc(12ull * 1024 * 1024);
    bf16* atA  = (bf16*)alloc(2ull * 1024 * 1024);
    bf16* atB  = (bf16*)alloc(2ull * 1024 * 1024);
    bf16* tbuf = (bf16*)alloc(512ull * 1096 * 2);
    bf16* ybuf = (bf16*)alloc(512ull * 1096 * 2);
    bf16* qb   = (bf16*)alloc(512ull * 1096 * 2);
    bf16* kb_  = (bf16*)alloc(512ull * 1096 * 2);
    bf16* vb   = (bf16*)alloc(512ull * 1096 * 2);
    bf16* qh   = (bf16*)alloc(4ull * 512 * 274 * 2);
    bf16* kh   = (bf16*)alloc(4ull * 512 * 274 * 2);
    bf16* vh   = (bf16*)alloc(4ull * 512 * 274 * 2);
    bf16* projT= (bf16*)alloc(274ull * 256 * 2);
    bf16* qp   = (bf16*)alloc(4ull * 512 * 256 * 2);
    bf16* kpT  = (bf16*)alloc(4ull * 256 * 512 * 2);
    bf16* kvb  = (bf16*)alloc(4ull * 256 * 274 * 2);
    float* ksum  = (float*)alloc(4ull * 256 * 4);
    float* denom = (float*)alloc(4ull * 512 * 4);
    bf16* ohb  = (bf16*)alloc(4ull * 512 * 274 * 2);
    bf16* mrg  = (bf16*)alloc(512ull * 1096 * 2);
    bf16* h1   = (bf16*)alloc(512ull * 4384 * 2);
    bf16* fb   = (bf16*)alloc(512ull * 2048 * 2);
    float* mo  = (float*)alloc(64);
    if (off > ws_size) return;  // workspace too small -> deterministic no-op

    auto EW = [](long long n) { return dim3((unsigned)((n + 255) / 256)); };
    auto cvt = [&](const float* s, bf16* d, long long n) {
        gf_f32_to_bf16<<<EW(n), 256, 0, stream>>>(s, d, n);
    };
    auto gemm = [&](const bf16* A, int lda, const bf16* B, int ldb,
                    const float* bias, const bf16* resid, int ldr,
                    bf16* C, int ldc, int M, int N, int K,
                    int convW, int cin, float alpha, int act, int storeT) {
        dim3 g((unsigned)((N + 63) / 64), (unsigned)((M + 127) / 128));
        gf_wmma_gemm<<<g, 256, 0, stream>>>(A, lda, B, ldb, bias, resid, ldr,
                                            C, ldc, M, N, K, convW, cin, alpha, act, storeT);
    };
    auto gemmW = [&](const bf16* A, int lda, const float* Wf, const float* bias,
                     const bf16* resid, int ldr, bf16* C, int ldc,
                     int M, int N, int K, float alpha, int act) {
        cvt(Wf, wscr, (long long)K * N);
        gemm(A, lda, wscr, N, bias, resid, ldr, C, ldc, M, N, K, 1, K, alpha, act, 0);
    };

    struct CBp { const float *g, *b, *m, *v, *w, *bias; };
    auto readCB = [&]() { CBp c; c.g = nx(); c.b = nx(); c.m = nx(); c.v = nx(); c.w = nx(); c.bias = nx(); return c; };
    auto convblock = [&](const bf16* x, bf16* tb, bf16* outp, const bf16* resid,
                         int L, int ci, int co, int W, const CBp& p) {
        long long n = (long long)L * ci;
        gf_bn_gelu<<<EW(n), 256, 0, stream>>>(x, p.g, p.b, p.m, p.v, tb, n, ci);
        cvt(p.w, wscr, (long long)W * ci * co);
        gemm(tb, ci, wscr, co, p.bias, resid, co, outp, co, L, co, W * ci, W, ci, 1.0f, 0, 0);
    };

    // ================= sequence tower =================
    const float* stem_w = nx(); const float* stem_b = nx();
    gf_stem_conv<<<EW(65536LL * 512), 256, 0, stream>>>(seq, stem_w, stem_b, bA, 65536, 4, 512, 15);
    CBp sres = readCB();
    convblock(bA, bB, bA, bA, 65536, 512, 512, 1, sres);           // x += convblock(x)
    gf_maxpool<<<EW(32768LL * 512), 256, 0, stream>>>(bA, bB, 32768, 512, 2);
    bf16* cur = bB; bf16* tmp = bA;
    int L = 32768, ci = 512;
    const int F[6] = {512, 512, 768, 768, 1024, 1024};
    for (int bi = 0; bi < 6; ++bi) {
        CBp cb = readCB(); CBp rs = readCB();
        int f = F[bi];
        convblock(cur, tmp, cur, nullptr, L, ci, f, 5, cb);        // x = convblock(x)
        convblock(cur, tmp, cur, cur,     L, f,  f, 1, rs);        // x += convblock(x)
        gf_maxpool<<<EW((long long)(L / 2) * f), 256, 0, stream>>>(cur, tmp, L / 2, f, 2);
        bf16* s = cur; cur = tmp; tmp = s;
        L /= 2; ci = f;
    }
    bf16* xseq = cur;  // [512, 1024]

    // ================= ATAC tower =================
    const float* sw_a = nx(); const float* sb_a = nx();
    gf_stem_conv<<<EW(16384LL * 32), 256, 0, stream>>>(atac, sw_a, sb_a, atA, 16384, 1, 32, 50);
    CBp sres_a = readCB();
    convblock(atA, atB, atA, atA, 16384, 32, 32, 1, sres_a);
    gf_maxpool<<<EW(8192LL * 32), 256, 0, stream>>>(atA, atB, 8192, 32, 2);
    bf16* ac = atB; bf16* at = atA;
    int La = 8192, ca = 32;
    const int Fa[2] = {32, 64};
    for (int bi = 0; bi < 2; ++bi) {
        CBp cb = readCB(); CBp rs = readCB();
        int f = Fa[bi];
        convblock(ac, at, ac, nullptr, La, ca, f, 5, cb);
        convblock(ac, at, ac, ac,      La, f,  f, 1, rs);
        gf_maxpool<<<EW((long long)(La / 4) * f), 256, 0, stream>>>(ac, at, La / 4, f, 4);
        bf16* s = ac; ac = at; at = s;
        La /= 4; ca = f;
    }
    bf16* aseq = ac;  // [512, 64]

    // ================= motif MLP + concat =================
    const float* mw1 = nx(); const float* mb1 = nx();
    const float* mw2 = nx(); const float* mb2 = nx();
    gf_motif<<<1, 64, 0, stream>>>(mot, mw1, mb1, mw2, mb2, mo);
    gf_concat<<<EW(512LL * 1096), 256, 0, stream>>>(xseq, aseq, mo, tbuf);

    // ================= 8 Performer layers =================
    for (int li = 0; li < 8; ++li) {
        const float *ln1g = nx(), *ln1b = nx();
        const float *wq = nx(), *bq = nx(), *wk = nx(), *bk = nx();
        const float *wv = nx(), *bv = nx(), *wo = nx(), *bo = nx();
        const float *proj = nx();
        const float *ln2g = nx(), *ln2b = nx();
        const float *w1 = nx(), *b1 = nx(), *w2 = nx(), *b2 = nx();

        gf_ln<<<512, 256, 0, stream>>>(tbuf, ln1g, ln1b, ybuf, 1096);
        gemmW(ybuf, 1096, wq, bq, nullptr, 0, qb,  1096, 512, 1096, 1096, 1.0f, 0);
        gemmW(ybuf, 1096, wk, bk, nullptr, 0, kb_, 1096, 512, 1096, 1096, 1.0f, 0);
        gemmW(ybuf, 1096, wv, bv, nullptr, 0, vb,  1096, 512, 1096, 1096, 1.0f, 0);
        gf_rotary_split<<<EW(512LL * 4 * 137), 256, 0, stream>>>(qb,  qh, 512);
        gf_rotary_split<<<EW(512LL * 4 * 137), 256, 0, stream>>>(kb_, kh, 512);
        gf_split_heads<<<EW(512LL * 4 * 274), 256, 0, stream>>>(vb, vh, 512);
        gf_transpose_f32_bf16<<<EW(256LL * 274), 256, 0, stream>>>(proj, projT, 256, 274);

        for (int h = 0; h < 4; ++h) {
            const bf16* qhh = qh + (size_t)h * 512 * 274;
            const bf16* khh = kh + (size_t)h * 512 * 274;
            const bf16* vhh = vh + (size_t)h * 512 * 274;
            // qp = relu((1/16) * q . projT) + STAB      [512,256]
            gemm(qhh, 274, projT, 256, nullptr, nullptr, 0,
                 qp + (size_t)h * 512 * 256, 256, 512, 256, 274, 1, 274, 0.0625f, 2, 0);
            // kpT = (relu((1/16) * k . projT) + STAB)^T [256,512] via storeT
            gemm(khh, 274, projT, 256, nullptr, nullptr, 0,
                 kpT + (size_t)h * 256 * 512, 512, 512, 256, 274, 1, 274, 0.0625f, 2, 1);
            // kv = kpT @ v                               [256,274]
            gemm(kpT + (size_t)h * 256 * 512, 512, vhh, 274, nullptr, nullptr, 0,
                 kvb + (size_t)h * 256 * 274, 274, 256, 274, 512, 1, 512, 1.0f, 0, 0);
        }
        gf_rowsum<<<EW(1024), 256, 0, stream>>>(kpT, ksum);
        gf_denom<<<EW(2048), 256, 0, stream>>>(qp, ksum, denom);
        for (int h = 0; h < 4; ++h) {
            gemm(qp + (size_t)h * 512 * 256, 256, kvb + (size_t)h * 256 * 274, 274,
                 nullptr, nullptr, 0, ohb + (size_t)h * 512 * 274, 274,
                 512, 274, 256, 1, 256, 1.0f, 0, 0);
        }
        gf_merge<<<EW(512LL * 1096), 256, 0, stream>>>(ohb, denom, mrg);
        // t = t + merged @ wo + bo
        gemmW(mrg, 1096, wo, bo, tbuf, 1096, tbuf, 1096, 512, 1096, 1096, 1.0f, 0);
        // FFN
        gf_ln<<<512, 256, 0, stream>>>(tbuf, ln2g, ln2b, ybuf, 1096);
        gemmW(ybuf, 1096, w1, b1, nullptr, 0, h1, 4384, 512, 4384, 1096, 1.0f, 1);
        gemmW(h1, 4384, w2, b2, tbuf, 1096, tbuf, 1096, 512, 1096, 4384, 1.0f, 0);
    }

    // ================= output head =================
    const float* lnfg = nx(); const float* lnfb = nx();
    CBp fp = readCB();
    gf_ln<<<512, 256, 0, stream>>>(tbuf, lnfg, lnfb, ybuf, 1096);
    {
        long long n = 512LL * 1096;
        gf_bn_gelu<<<EW(n), 256, 0, stream>>>(ybuf, fp.g, fp.b, fp.m, fp.v, mrg, n, 1096);
    }
    // conv 1x1 (1096->2048) with the outer GELU fused into the epilogue
    gemmW(mrg, 1096, fp.w, fp.bias, nullptr, 0, fb, 2048, 512, 2048, 1096, 1.0f, 1);
    const float* pw = nx(); const float* pb = nx();
    gf_final<<<508, 256, 0, stream>>>(fb, pw, pb, (float*)d_out);
}
